// NerveNetGNN_47201690583597
// MI455X (gfx1250) — matrix operand order repaired and checked by reading, not compiled, not run on hardware
//
#include <hip/hip_runtime.h>
#include <stdint.h>

#define BATCH   32768
#define BT      16          // samples per workgroup
#define NTHREADS 256
#define NWAVES  8

typedef _Float16 v16h __attribute__((ext_vector_type(16)));
typedef _Float16 v8h  __attribute__((ext_vector_type(8)));
typedef float    v8f  __attribute__((ext_vector_type(8)));

union F16x16 { v16h v; v8h h[2]; };

// LDS partition (bytes). Activations use padded rows: sample stride = 16 rows x 64 f16.
#define OFF_WFRAG 0          // 5 * 4096 f16 = 40960 B (gp1,gp2,gv1,gv2,pol1 fragments)
#define OFF_BUFX  40960      // 16*16*64 f16 = 32768 B (node embeddings, node-padded)
#define OFF_BUFT  73728      // 16*64*16 f16 = 32768 B (t, feature-major, node fast axis)
#define OFF_BUFC  106496     // 16*16*64 f16 = 32768 B (current activation, node-padded)
#define OFF_ADJ   139264     // 16*32 f16 = 1024 B (padded adjacency)
#define OFF_OBS   140288     // 16*28 f32 = 1792 B
#define SMEM_BYTES 142080    // x2 workgroups = 278KB < 320KB/WGP

// gfx1250 hardware tanh (TRANS op, co-executes with WMMA); guarded fallback.
__device__ __forceinline__ float fast_tanh(float x) {
#if __has_builtin(__builtin_amdgcn_tanhf)
  return __builtin_amdgcn_tanhf(x);
#elif __has_builtin(__builtin_amdgcn_tanh_f32)
  return __builtin_amdgcn_tanh_f32(x);
#else
  return tanhf(x);
#endif
}

__device__ __forceinline__ v8f wmma16(v16h a, v16h b, v8f c) {
  return __builtin_amdgcn_wmma_f32_16x16x32_f16(false, a, false, b, (short)0, c,
                                                false, false);
}

__device__ __forceinline__ v8f splat8(float x) {
  v8f c;
#pragma unroll
  for (int i = 0; i < 8; ++i) c[i] = x;
  return c;
}

// A fragment (16x32 f16) from a row-major LDS row. Contiguous K layout:
// lanes 0-15 hold K {0..15}, lanes 16-31 hold K {16..31}.
__device__ __forceinline__ v16h load_afrag(const _Float16* rowb, int lh, int kc) {
  F16x16 u;
  const _Float16* p = rowb + kc * 32 + lh * 16;
  u.h[0] = *(const v8h*)(p);
  u.h[1] = *(const v8h*)(p + 8);
  return u.v;
}

// B fragment (32x16 f16) from pre-shuffled weight fragments.
__device__ __forceinline__ v16h load_bfrag(const _Float16* wf, int kc, int nt, int lane) {
  F16x16 u;
  const _Float16* p = wf + (((kc * 4 + nt) * 32) + lane) * 16;
  u.h[0] = *(const v8h*)(p);
  u.h[1] = *(const v8h*)(p + 8);
  return u.v;
}

// Shuffle one 64x64 f32 row-major weight into B-fragment layout (f16) in LDS.
__device__ __forceinline__ void xform_weight(const float* __restrict__ W,
                                             _Float16* dst, int tid) {
  int kc   = tid >> 7;
  int rem  = tid & 127;
  int nt   = rem >> 5;
  int code = rem & 31;              // khalf*16 + nn  (== lane code)
  int n     = nt * 16 + (code & 15);
  int kbase = kc * 32 + (code >> 4) * 16;
  F16x16 u;
#pragma unroll
  for (int kk = 0; kk < 16; ++kk) u.v[kk] = (_Float16)W[(kbase + kk) * 64 + n];
  v8h* p = (v8h*)(dst + tid * 16);
  p[0] = u.h[0];
  p[1] = u.h[1];
}

// Pack 8 f32 D-values to f16 and store as ONE b128 (node is bufT's fast axis).
// Pad rows 12..15 of the source are zero, so the pad slots are rewritten with
// exact zeros every layer - no separate pad-clear pass needed.
__device__ __forceinline__ void store_T_packed(v8f c, int mt, int nt, _Float16* bufT,
                                               int lp, int lh) {
  v8h o;
#pragma unroll
  for (int rr = 0; rr < 8; ++rr) o[rr] = (_Float16)c[rr];
  *(v8h*)(bufT + (mt * 64 + nt * 16 + lp) * 16 + 8 * lh) = o;
}

// t = src @ W per sample tile (16 samples x 16 padded node rows).
// Two independent WMMA chains per wave iteration fill the f16 hazard slots.
__device__ __forceinline__ void gemm_hw_to_T(const _Float16* src, const _Float16* wf,
                                             _Float16* bufT, int tid) {
  int lane = tid & 31, wave = tid >> 5;
  int lp = lane & 15, lh = lane >> 4;
  for (int tb = wave; tb < 64; tb += 2 * NWAVES) {       // tasks tb and tb+8
    int t1 = tb + NWAVES;
    int mt0 = tb >> 2, nt0 = tb & 3;
    int mt1 = t1 >> 2, nt1 = t1 & 3;
    const _Float16* row0 = src + (mt0 * 16 + lp) * 64;   // padded row = b*16 + node
    const _Float16* row1 = src + (mt1 * 16 + lp) * 64;
    v8f c0 = splat8(0.f), c1 = splat8(0.f);
#pragma unroll
    for (int kc = 0; kc < 2; ++kc) {
      v16h a0 = load_afrag(row0, lh, kc);
      v16h b0 = load_bfrag(wf, kc, nt0, lane);
      v16h a1 = load_afrag(row1, lh, kc);
      v16h b1 = load_bfrag(wf, kc, nt1, lane);
      c0 = wmma16(a0, b0, c0);
      c1 = wmma16(a1, b1, c1);
    }
    store_T_packed(c0, mt0, nt0, bufT, lp, lh);
    store_T_packed(c1, mt1, nt1, bufT, lp, lh);
  }
}

__device__ __forceinline__ v16h load_adjB(const _Float16* bufT, int b, int nt,
                                          int lp, int lh) {
  F16x16 u;
#pragma unroll
  for (int i = 0; i < 16; ++i) u.v[i] = (_Float16)0.f;
  if (lh == 0) {                                         // K rows 0..15 (12 real + 0-pad)
    const _Float16* p = bufT + (b * 64 + nt * 16 + lp) * 16;
    u.h[0] = *(const v8h*)(p);
    u.h[1] = *(const v8h*)(p + 8);
  }                                                      // K rows 16..31: zeros
  return u.v;
}

__device__ __forceinline__ void store_adjD(v8f c, _Float16* dst, int b, int nt,
                                           int lp, int lh) {
#pragma unroll
  for (int rr = 0; rr < 8; ++rr) {
    int m = rr + 8 * lh;                                 // node
    if (m < 12)
      dst[(b * 16 + m) * 64 + nt * 16 + lp] = (_Float16)fast_tanh(c[rr]);
  }
}

// dst[b,n,d] = tanh( sum_m adj[n,m]*t[b,m,d] + bias[d] ) via per-sample WMMA.
__device__ __forceinline__ void adj_stage(v16h adjA, const _Float16* bufT,
                                          _Float16* dst, const float* __restrict__ bias,
                                          int tid) {
  int lane = tid & 31;
  int lp = lane & 15, lh = lane >> 4;
  for (int tb = (tid >> 5); tb < BT * 4; tb += 2 * NWAVES) {  // tasks tb, tb+8
    int b0 = tb >> 2, nt0 = tb & 3;
    int b1 = (tb + NWAVES) >> 2, nt1 = (tb + NWAVES) & 3;
    v16h u0 = load_adjB(bufT, b0, nt0, lp, lh);
    v16h u1 = load_adjB(bufT, b1, nt1, lp, lh);
    v8f c0 = splat8(bias[nt0 * 16 + lp]);                // bias enters via C
    v8f c1 = splat8(bias[nt1 * 16 + lp]);
    c0 = wmma16(adjA, u0, c0);
    c1 = wmma16(adjA, u1, c1);
    store_adjD(c0, dst, b0, nt0, lp, lh);
    store_adjD(c1, dst, b1, nt1, lp, lh);
  }
}

__global__ void __launch_bounds__(NTHREADS)
nervenet_fused(const float* __restrict__ obs,    const float* __restrict__ adjn,
               const float* __restrict__ w_in_t, const float* __restrict__ b_in_t,
               const float* __restrict__ w_in_j, const float* __restrict__ b_in_j,
               const float* __restrict__ w_gp1,  const float* __restrict__ b_gp1,
               const float* __restrict__ w_gp2,  const float* __restrict__ b_gp2,
               const float* __restrict__ w_gv1,  const float* __restrict__ b_gv1,
               const float* __restrict__ w_gv2,  const float* __restrict__ b_gv2,
               const float* __restrict__ w_pol1, const float* __restrict__ b_pol1,
               const float* __restrict__ w_pol2, const float* __restrict__ b_pol2,
               const float* __restrict__ w_val1, const float* __restrict__ b_val1,
               const float* __restrict__ w_val2, const float* __restrict__ b_val2,
               float* __restrict__ out) {
  extern __shared__ char smem[];
  _Float16* wfrag  = (_Float16*)(smem + OFF_WFRAG);
  _Float16* bufX   = (_Float16*)(smem + OFF_BUFX);
  _Float16* bufT   = (_Float16*)(smem + OFF_BUFT);
  _Float16* bufC   = (_Float16*)(smem + OFF_BUFC);
  _Float16* adjPad = (_Float16*)(smem + OFF_ADJ);
  float*    lobs   = (float*)(smem + OFF_OBS);

  const int tid  = threadIdx.x;
  const int lane = tid & 31;
  const int lp = lane & 15, lh = lane >> 4;
  const int b0 = blockIdx.x * BT;

  // ---- Phase 0: stage obs/adjacency/weight fragments; zero activation pad rows ----
  for (int e = tid; e < BT * 28; e += NTHREADS) lobs[e] = obs[b0 * 28 + e];
  for (int e = tid; e < 512; e += NTHREADS) {
    int m = e >> 5, k = e & 31;
    adjPad[e] = (m < 12 && k < 12) ? (_Float16)adjn[m * 12 + k] : (_Float16)0.f;
  }
  // Zero pad rows (node 12..15) of bufX/bufC once; nothing ever writes them again,
  // so stage-1 A-rows 12..15 are always exactly 0 (=> bufT pads stay 0 each layer).
  for (int e = tid; e < BT * 64; e += NTHREADS) {        // 64 u64 per sample per buffer
    int b = e >> 6, q = e & 63;
    ((uint64_t*)(bufX + b * 1024 + 768))[q] = 0ull;
    ((uint64_t*)(bufC + b * 1024 + 768))[q] = 0ull;
  }
  {
    const float* Ws[5] = {w_gp1, w_gp2, w_gv1, w_gv2, w_pol1};
#pragma unroll
    for (int w = 0; w < 5; ++w) xform_weight(Ws[w], wfrag + w * 4096, tid);
  }
  __syncthreads();

  // Adjacency A-fragment lives in registers for the whole kernel.
  v16h adjA = load_afrag(adjPad + lp * 32, lh, 0);

  // ---- Phase 1: node embeddings -> bufX (tanh(obs @ W_in + b)), packed stores ----
  // 256 padded rows; 4 threads per row, 16 features each; 4 passes.
  for (int base = 0; base < BT * 16; base += 64) {
    int r = base + (tid >> 2);
    int b = r >> 4, node = r & 15;
    if (node < 12) {
      int d0 = (tid & 3) * 16;
      const float* ob = lobs + b * 28;
      F16x16 u;
      if (node == 0) {
#pragma unroll
        for (int i = 0; i < 16; ++i) {
          int d = d0 + i;
          float s = b_in_t[d];
#pragma unroll
          for (int q = 0; q < 6; ++q) s += ob[q] * w_in_t[q * 64 + d];
          u.v[i] = (_Float16)fast_tanh(s);
        }
      } else {
        int j = node - 1;
        float o0 = ob[6 + 2 * j], o1 = ob[7 + 2 * j];
#pragma unroll
        for (int i = 0; i < 16; ++i) {
          int d = d0 + i;
          u.v[i] = (_Float16)fast_tanh(b_in_j[d] + o0 * w_in_j[d] + o1 * w_in_j[64 + d]);
        }
      }
      v8h* p = (v8h*)(bufX + r * 64 + d0);
      p[0] = u.h[0];
      p[1] = u.h[1];
    }
  }
  __syncthreads();

  // ---- Policy GCN layers ----
  gemm_hw_to_T(bufX, wfrag + 0 * 4096, bufT, tid); __syncthreads();
  adj_stage(adjA, bufT, bufC, b_gp1, tid);         __syncthreads();
  gemm_hw_to_T(bufC, wfrag + 1 * 4096, bufT, tid); __syncthreads();
  adj_stage(adjA, bufT, bufC, b_gp2, tid);         __syncthreads();

  // ---- Policy head: h = tanh(xp[:,1:,:] @ w_pol1 + b) -> bufT rows (b*16+j) ----
  // Per-sample tiles: 64 uniform tasks, paired for two independent chains.
  for (int tb = (tid >> 5); tb < 64; tb += 2 * NWAVES) {
    int t1 = tb + NWAVES;
    int mt0 = tb >> 2, nt0 = tb & 3;
    int mt1 = t1 >> 2, nt1 = t1 & 3;
    const _Float16* row0 = bufC + (mt0 * 16 + 1 + lp) * 64;  // A row j = lp -> node j+1
    const _Float16* row1 = bufC + (mt1 * 16 + 1 + lp) * 64;
    v8f c0 = splat8(b_pol1[nt0 * 16 + lp]);
    v8f c1 = splat8(b_pol1[nt1 * 16 + lp]);
#pragma unroll
    for (int kc = 0; kc < 2; ++kc) {
      c0 = wmma16(load_afrag(row0, lh, kc),
                  load_bfrag(wfrag + 4 * 4096, kc, nt0, lane), c0);
      c1 = wmma16(load_afrag(row1, lh, kc),
                  load_bfrag(wfrag + 4 * 4096, kc, nt1, lane), c1);
    }
#pragma unroll
    for (int rr = 0; rr < 8; ++rr) {
      int j = rr + 8 * lh;
      if (j < 11) {
        bufT[(mt0 * 16 + j) * 64 + nt0 * 16 + lp] = (_Float16)fast_tanh(c0[rr]);
        bufT[(mt1 * 16 + j) * 64 + nt1 * 16 + lp] = (_Float16)fast_tanh(c1[rr]);
      }
    }
  }
  __syncthreads();
  if (tid < BT * 11) {                                   // latent_pi = h @ w_pol2 + b
    int b = tid / 11, j = tid % 11;
    float s = b_pol2[0];
    for (int d = 0; d < 64; ++d) s += (float)bufT[(b * 16 + j) * 64 + d] * w_pol2[d];
    out[(b0 + b) * 11 + j] = s;
  }
  __syncthreads();

  // ---- Value GCN layers ----
  gemm_hw_to_T(bufX, wfrag + 2 * 4096, bufT, tid); __syncthreads();
  adj_stage(adjA, bufT, bufC, b_gv1, tid);         __syncthreads();
  gemm_hw_to_T(bufC, wfrag + 3 * 4096, bufT, tid); __syncthreads();
  adj_stage(adjA, bufT, bufC, b_gv2, tid);         __syncthreads();

  // ---- Value head: hv = tanh(flat(16x768) @ w_val1 + b); vf = hv @ w_val2 + b ----
  // flat rows stay contiguous: real nodes occupy first 768 f16 of each 1024-f16 block.
  float* fT = (float*)bufT;
  for (int t = (tid >> 5); t < 8; t += NWAVES) {
    int nt = t & 3, part = t >> 2;
    const _Float16* rowb = bufC + lp * 1024;             // A row = sample lp, K=768
    int n = nt * 16 + lp;
    v8f c0 = splat8(0.f), c1 = splat8(0.f);
    for (int kc = part * 12; kc < part * 12 + 12; kc += 2) {
#pragma unroll
      for (int half = 0; half < 2; ++half) {
        int kcc = kc + half;
        v16h a = load_afrag(rowb, lh, kcc);
        F16x16 ub;
#pragma unroll
        for (int kk = 0; kk < 16; ++kk) {
          int k = kcc * 32 + lh * 16 + kk;
          ub.v[kk] = (_Float16)w_val1[k * 64 + n];       // L2-resident gather
        }
        if (half == 0) c0 = wmma16(a, ub.v, c0);
        else           c1 = wmma16(a, ub.v, c1);
      }
    }
    c0 = c0 + c1;
#pragma unroll
    for (int rr = 0; rr < 8; ++rr)
      fT[part * 1024 + (rr + 8 * lh) * 64 + n] = c0[rr];
  }
  __syncthreads();
  for (int e = tid; e < BT * 64; e += NTHREADS) {        // combine halves + bias + tanh
    int d = e & 63;
    fT[2048 + e] = fast_tanh(fT[e] + fT[1024 + e] + b_val1[d]);
  }
  __syncthreads();
  if (tid < BT) {
    float s = b_val2[0];
    for (int d = 0; d < 64; ++d) s += fT[2048 + tid * 64 + d] * w_val2[d];
    out[BATCH * 11 + b0 + tid] = s;                      // latent_vf
  }
}

extern "C" void kernel_launch(void* const* d_in, const int* in_sizes, int n_in,
                              void* d_out, int out_size, void* d_ws, size_t ws_size,
                              hipStream_t stream) {
  (void)in_sizes; (void)n_in; (void)d_ws; (void)ws_size; (void)out_size;
  // Idempotent + deterministic (not a stream op): safe under graph capture.
  hipFuncSetAttribute((const void*)nervenet_fused,
                      hipFuncAttributeMaxDynamicSharedMemorySize, SMEM_BYTES);
  nervenet_fused<<<BATCH / BT, NTHREADS, SMEM_BYTES, stream>>>(
      (const float*)d_in[0],  (const float*)d_in[1],
      (const float*)d_in[2],  (const float*)d_in[3],
      (const float*)d_in[4],  (const float*)d_in[5],
      (const float*)d_in[6],  (const float*)d_in[7],
      (const float*)d_in[8],  (const float*)d_in[9],
      (const float*)d_in[10], (const float*)d_in[11],
      (const float*)d_in[12], (const float*)d_in[13],
      (const float*)d_in[14], (const float*)d_in[15],
      (const float*)d_in[16], (const float*)d_in[17],
      (const float*)d_in[18], (const float*)d_in[19],
      (const float*)d_in[20], (const float*)d_in[21],
      (float*)d_out);
}
